// GrapherModule_72945724555891
// MI455X (gfx1250) — matrix operand order repaired and checked by reading, not compiled, not run on hardware
//
#include <hip/hip_runtime.h>
#include <math.h>

typedef _Float16 h16;
typedef __attribute__((ext_vector_type(16))) _Float16 v16h;
typedef __attribute__((ext_vector_type(8)))  _Float16 v8h;
typedef __attribute__((ext_vector_type(8)))  float    v8f;

union V16H { v16h v; v8h h[2]; };

// ---------------- WMMA helpers (CDNA5 wave32) ----------------

__device__ inline v8f wmma_f16(v16h a, v16h b, v8f c) {
  // v_wmma_f32_16x16x32_f16: D = A(16x32,f16) * B(32x16,f16) + C(16x16,f32)
  return __builtin_amdgcn_wmma_f32_16x16x32_f16(
      /*neg_a=*/false, a, /*neg_b=*/false, b,
      /*c_mod=*/(short)0, c, /*reuse_a=*/false, /*reuse_b=*/false);
}

// A fragment: 16x32 f16 tile from row-major matrix, A points at (m0,k0).
// Lane m (0..15) holds K = {0..7, 16..23}; lane m+16 holds K = {8..15, 24..31}.
__device__ inline v16h load_a_frag(const h16* __restrict__ A, int lda, int lane) {
  int m  = lane & 15;
  int kb = (lane >> 4) << 3;   // 0 or 8
  V16H r;
  r.h[0] = *(const v8h*)(A + (size_t)m * lda + kb);
  r.h[1] = *(const v8h*)(A + (size_t)m * lda + kb + 16);
  return r.v;
}

// B fragment: 32x16 f16 tile from row-major KxN matrix, Bp points at (k0,n0).
// Lane n (0..15) holds column n, K = {0..7, 16..23}; lane n+16 holds K = {8..15, 24..31}.
__device__ inline v16h load_b_frag(const h16* __restrict__ Bp, int ldb, int lane) {
  int n  = lane & 15;
  int kb = (lane >> 4) << 3;   // 0 or 8
  const h16* p = Bp + n;
  v16h r;
#pragma unroll
  for (int s = 0; s < 8; ++s) {
    r[s]     = p[(size_t)(kb + s) * ldb];
    r[s + 8] = p[(size_t)(kb + 16 + s) * ldb];
  }
  return r;
}

// ---------------- elementwise convert ----------------

__global__ void f32_to_f16_kernel(const float* __restrict__ in, h16* __restrict__ out, int n) {
  int i = blockIdx.x * blockDim.x + threadIdx.x;
  if (i < n) out[i] = (h16)in[i];
}

// ---------------- GEMM (M x KTOT) @ (KTOT x 128) + bias, fused BN stats ----------------
// One wave computes a 16x128 row strip; A fragments reused across all 8 N tiles.

template <int KTOT>
__global__ void gemm_bias_stats_kernel(const h16* __restrict__ A,
                                       const h16* __restrict__ Bw,
                                       const float* __restrict__ bias,
                                       float* __restrict__ out,
                                       float* __restrict__ colsum,
                                       float* __restrict__ colsq,
                                       int M) {
  constexpr int KC = KTOT / 32;
  int lane = threadIdx.x & 31;
  int wave = threadIdx.x >> 5;
  int m0 = (blockIdx.x * (blockDim.x >> 5) + wave) * 16;
  if (m0 >= M) return;

  v16h afr[KC];
#pragma unroll
  for (int kc = 0; kc < KC; ++kc)
    afr[kc] = load_a_frag(A + (size_t)m0 * KTOT + kc * 32, KTOT, lane);

  v8f acc[8];
#pragma unroll
  for (int nt = 0; nt < 8; ++nt) acc[nt] = {};

#pragma unroll
  for (int nt = 0; nt < 8; ++nt) {
#pragma unroll
    for (int kc = 0; kc < KC; ++kc) {
      v16h bfr = load_b_frag(Bw + (size_t)(kc * 32) * 128 + nt * 16, 128, lane);
      acc[nt] = wmma_f16(afr[kc], bfr, acc[nt]);
    }
  }

  // Epilogue: bias add, store fp32, per-column partial sums for BN stats.
  int nlo  = lane & 15;
  int mrow = m0 + ((lane >> 4) << 3);
#pragma unroll
  for (int nt = 0; nt < 8; ++nt) {
    int col  = nt * 16 + nlo;
    float bv = bias[col];
    float s = 0.f, sq = 0.f;
#pragma unroll
    for (int i = 0; i < 8; ++i) {
      float v = acc[nt][i] + bv;
      out[(size_t)(mrow + i) * 128 + col] = v;
      s += v; sq += v * v;
    }
    s  += __shfl_xor(s, 16, 32);
    sq += __shfl_xor(sq, 16, 32);
    if (lane < 16) {
      atomicAdd(&colsum[col], s);
      atomicAdd(&colsq[col], sq);
    }
  }
}

// ---------------- BN stat finalize ----------------

__global__ void bn_finalize_kernel(const float* __restrict__ sum, const float* __restrict__ sq,
                                   float* __restrict__ mean, float* __restrict__ rstd, float invM) {
  int c = threadIdx.x;
  float m = sum[c] * invM;
  float v = sq[c] * invM - m * m;
  mean[c] = m;
  rstd[c] = rsqrtf(v + 1e-5f);
}

// ---------------- BN apply after fc1: produce h(f32), h(f16), hT(f16), row |h|^2 ----------------

__global__ void bn_apply_h_kernel(const float* __restrict__ pre,
                                  const float* __restrict__ mean, const float* __restrict__ rstd,
                                  const float* __restrict__ g, const float* __restrict__ be,
                                  float* __restrict__ h, h16* __restrict__ hh,
                                  h16* __restrict__ hT, float* __restrict__ sqv) {
  int row = blockIdx.x;       // 0..32767
  int c   = threadIdx.x;      // 0..127
  int b   = row >> 12;
  int n   = row & 4095;
  float v = g[c] * (pre[(size_t)row * 128 + c] - mean[c]) * rstd[c] + be[c];
  h[(size_t)row * 128 + c] = v;
  h16 hv = (h16)v;
  hh[(size_t)row * 128 + c] = hv;
  hT[((size_t)b * 128 + c) * 4096 + n] = hv;
  __shared__ float red[128];
  red[c] = v * v;
  __syncthreads();
  for (int s = 64; s > 0; s >>= 1) {
    if (c < s) red[c] += red[c + s];
    __syncthreads();
  }
  if (c == 0) sqv[row] = red[0];
}

// ---------------- KNN helpers: statically-indexed prefetch + compute/select ----------------

__device__ __forceinline__ void knn_prefetch(const h16* __restrict__ hb,
                                             const float* __restrict__ sqb,
                                             int nt, int lane, int mm,
                                             v16h (&afr)[4], float4& sqa, float4& sqc) {
  int n0 = nt * 16;
#pragma unroll
  for (int kc = 0; kc < 4; ++kc)
    afr[kc] = load_a_frag(hb + (size_t)n0 * 128 + kc * 32, 128, lane);
  sqa = *(const float4*)(sqb + n0 + mm);
  sqc = *(const float4*)(sqb + n0 + mm + 4);
}

__device__ __forceinline__ void knn_compute(int nt, int mm,
                                            const v16h (&afr)[4], const v16h (&qfr)[4],
                                            float4 sqa, float4 sqc,
                                            float (&best)[9], int (&bidx)[9]) {
  v8f acc = {};
#pragma unroll
  for (int kc = 0; kc < 4; ++kc)
    acc = wmma_f16(afr[kc], qfr[kc], acc);
  int cbase = nt * 16 + mm;
  float sqr[8] = { sqa.x, sqa.y, sqa.z, sqa.w, sqc.x, sqc.y, sqc.z, sqc.w };
#pragma unroll
  for (int i = 0; i < 8; ++i) {
    float d = sqr[i] - 2.f * acc[i];
    if (d < best[8]) {
      best[8] = d; bidx[8] = cbase + i;
#pragma unroll
      for (int p = 8; p > 0; --p) {
        if (best[p] < best[p - 1]) {
          float tf = best[p]; best[p] = best[p - 1]; best[p - 1] = tf;
          int   ti = bidx[p]; bidx[p] = bidx[p - 1]; bidx[p - 1] = ti;
        }
      }
    }
  }
}

// ---------------- KNN: streaming candidates @ queries^T (WMMA) + per-row top-9 ----------------
// Operand-swapped: A = 16 candidate rows (per tile), B = 16 query columns (fixed per wave).
// D[cand][query]: each lane owns query (lane&15) and 8 candidates in the accumulator
// => register-resident selection on all 32 lanes, no LDS/barriers in the loop.
// Double-buffered with STATIC buffer names (2x-unrolled loop) so fragments stay in VGPRs.
// Rank by (sq[cand] - 2*dot); sq[query] is row-constant and drops out of ordering.

__global__ void knn_topk_kernel(const h16* __restrict__ hh,   // [B*N,128]
                                const h16* __restrict__ hT,   // [B,128,4096]
                                const float* __restrict__ sqv,// [B*N]
                                int* __restrict__ idx) {      // [B*N,9]
  int lane = threadIdx.x & 31;
  int wave = threadIdx.x >> 5;
  int tile = blockIdx.x * 4 + wave;   // 2048 tiles total
  int b    = tile >> 8;               // 256 tiles per batch
  int m0   = (tile & 255) * 16;       // query tile base

  const h16* hb    = hh  + (size_t)b * 4096 * 128;
  const h16* tb    = hT  + (size_t)b * 128 * 4096;
  const float* sqb = sqv + (size_t)b * 4096;

  // Query B-fragments: fixed for the whole wave, reused 256x.
  v16h qfr[4];
#pragma unroll
  for (int kc = 0; kc < 4; ++kc)
    qfr[kc] = load_b_frag(tb + (size_t)(kc * 32) * 4096 + m0, 4096, lane);

  int mm = (lane >> 4) << 3;          // candidate octet base within tile: 0 or 8

  float best[9]; int bidx[9];
#pragma unroll
  for (int k = 0; k < 9; ++k) { best[k] = 3.4e38f; bidx[k] = 0; }

  // Two statically-named buffers; loop unrolled by 2 so all indices are constants.
  v16h  afr0[4], afr1[4];
  float4 sa0, sc0, sa1, sc1;
  knn_prefetch(hb, sqb, 0, lane, mm, afr0, sa0, sc0);

  for (int nt = 0; nt < 256; nt += 2) {
    knn_prefetch(hb, sqb, (nt + 1) & 255, lane, mm, afr1, sa1, sc1);
    knn_compute(nt, mm, afr0, qfr, sa0, sc0, best, bidx);
    knn_prefetch(hb, sqb, (nt + 2) & 255, lane, mm, afr0, sa0, sc0);
    knn_compute(nt + 1, mm, afr1, qfr, sa1, sc1, best, bidx);
  }

  // Merge the two half-lane lists (same query, disjoint candidate octets).
  __shared__ float mD[4][16][9];
  __shared__ int   mI[4][16][9];
  if (lane >= 16) {
#pragma unroll
    for (int k = 0; k < 9; ++k) { mD[wave][lane - 16][k] = best[k]; mI[wave][lane - 16][k] = bidx[k]; }
  }
  __syncthreads();
  if (lane < 16) {
    for (int k = 0; k < 9; ++k) {
      float d = mD[wave][lane][k];
      if (d >= best[8]) break;          // incoming list is sorted: rest also fail
      int ci = mI[wave][lane][k];
      best[8] = d; bidx[8] = ci;
#pragma unroll
      for (int p = 8; p > 0; --p) {
        if (best[p] < best[p - 1]) {
          float tf = best[p]; best[p] = best[p - 1]; best[p - 1] = tf;
          int   ti = bidx[p]; bidx[p] = bidx[p - 1]; bidx[p - 1] = ti;
        }
      }
    }
    size_t row = (size_t)b * 4096 + m0 + lane;
#pragma unroll
    for (int k = 0; k < 9; ++k) idx[row * 9 + k] = bidx[k];
  }
}

// ---------------- gather + max-relative aggregation + concat (f16) ----------------

__global__ void gather_cat_kernel(const float* __restrict__ h, const int* __restrict__ idx,
                                  h16* __restrict__ cat) {
  int row = blockIdx.x;
  int c   = threadIdx.x;
  int b   = row >> 12;
  __shared__ int nb[9];
  if (c < 9) nb[c] = idx[(size_t)row * 9 + c];
  __syncthreads();
  float hv = h[(size_t)row * 128 + c];
  const float* hb = h + (size_t)b * 4096 * 128;
  float mx = -3.4e38f;
#pragma unroll
  for (int k = 0; k < 9; ++k) mx = fmaxf(mx, hb[(size_t)nb[k] * 128 + c]);
  cat[(size_t)row * 256 + c]       = (h16)hv;
  cat[(size_t)row * 256 + 128 + c] = (h16)(mx - hv);
}

// ---------------- BN + exact GELU -> f16 ----------------

__global__ void bn_gelu_kernel(const float* __restrict__ pre,
                               const float* __restrict__ mean, const float* __restrict__ rstd,
                               const float* __restrict__ g, const float* __restrict__ be,
                               h16* __restrict__ out) {
  int i = blockIdx.x * blockDim.x + threadIdx.x;
  int c = i & 127;
  float v = g[c] * (pre[i] - mean[c]) * rstd[c] + be[c];
  float gel = 0.5f * v * (1.f + erff(v * 0.70710678118654752f));
  out[i] = (h16)gel;
}

// ---------------- final BN + residual ----------------

__global__ void bn_res_kernel(const float* __restrict__ pre,
                              const float* __restrict__ mean, const float* __restrict__ rstd,
                              const float* __restrict__ g, const float* __restrict__ be,
                              const float* __restrict__ x, float* __restrict__ out) {
  int i = blockIdx.x * blockDim.x + threadIdx.x;
  int c = i & 127;
  out[i] = g[c] * (pre[i] - mean[c]) * rstd[c] + be[c] + x[i];
}

// ---------------- launch ----------------

extern "C" void kernel_launch(void* const* d_in, const int* in_sizes, int n_in,
                              void* d_out, int out_size, void* d_ws, size_t ws_size,
                              hipStream_t stream) {
  (void)in_sizes; (void)n_in; (void)out_size; (void)ws_size;
  const float* x   = (const float*)d_in[0];
  const float* W1  = (const float*)d_in[1];
  const float* b1  = (const float*)d_in[2];
  const float* g1  = (const float*)d_in[3];
  const float* be1 = (const float*)d_in[4];
  const float* Wg  = (const float*)d_in[5];
  const float* bg  = (const float*)d_in[6];
  const float* gg  = (const float*)d_in[7];
  const float* beg = (const float*)d_in[8];
  const float* W2  = (const float*)d_in[9];
  const float* b2  = (const float*)d_in[10];
  const float* g2  = (const float*)d_in[11];
  const float* be2 = (const float*)d_in[12];

  constexpr int C = 128, N = 4096, Bb = 8, K = 9;
  constexpr int M = Bb * N;  // 32768

  char* ws = (char*)d_ws;
  size_t off = 0;
  auto alloc = [&](size_t bytes) -> void* {
    off = (off + 255) & ~(size_t)255;
    void* p = ws + off;
    off += bytes;
    return p;
  };

  h16*   xh   = (h16*)  alloc((size_t)M * C * 2);
  h16*   w1h  = (h16*)  alloc((size_t)C * C * 2);
  h16*   wgh  = (h16*)  alloc((size_t)2 * C * C * 2);
  h16*   w2h  = (h16*)  alloc((size_t)C * C * 2);
  float* pre  = (float*)alloc((size_t)M * C * 4);   // reused for all 3 pre-activations
  float* h    = (float*)alloc((size_t)M * C * 4);
  h16*   hh   = (h16*)  alloc((size_t)M * C * 2);
  h16*   hT   = (h16*)  alloc((size_t)M * C * 2);
  float* sqv  = (float*)alloc((size_t)M * 4);
  int*   idx  = (int*)  alloc((size_t)M * K * 4);
  h16*   cat  = (h16*)  alloc((size_t)M * 2 * C * 2);
  h16*   hgh  = (h16*)  alloc((size_t)M * C * 2);
  float* stats = (float*)alloc((size_t)12 * C * 4);

  float* sum1 = stats + 0 * C, *sq1 = stats + 1 * C, *mean1 = stats + 2 * C, *rstd1 = stats + 3 * C;
  float* sum2 = stats + 4 * C, *sq2 = stats + 5 * C, *mean2 = stats + 6 * C, *rstd2 = stats + 7 * C;
  float* sum3 = stats + 8 * C, *sq3 = stats + 9 * C, *mean3 = stats + 10 * C, *rstd3 = stats + 11 * C;

  hipMemsetAsync(stats, 0, 12 * C * sizeof(float), stream);

  f32_to_f16_kernel<<<(M * C + 255) / 256, 256, 0, stream>>>(x, xh, M * C);
  f32_to_f16_kernel<<<(C * C + 255) / 256, 256, 0, stream>>>(W1, w1h, C * C);
  f32_to_f16_kernel<<<(2 * C * C + 255) / 256, 256, 0, stream>>>(Wg, wgh, 2 * C * C);
  f32_to_f16_kernel<<<(C * C + 255) / 256, 256, 0, stream>>>(W2, w2h, C * C);

  // fc1 + stats
  gemm_bias_stats_kernel<128><<<M / 64, 128, 0, stream>>>(xh, w1h, b1, pre, sum1, sq1, M);
  bn_finalize_kernel<<<1, C, 0, stream>>>(sum1, sq1, mean1, rstd1, 1.f / M);
  bn_apply_h_kernel<<<M, C, 0, stream>>>(pre, mean1, rstd1, g1, be1, h, hh, hT, sqv);

  // KNN (streaming WMMA distance + register-resident top-9)
  knn_topk_kernel<<<M / 16 / 4, 128, 0, stream>>>(hh, hT, sqv, idx);

  // gather + max-relative + concat
  gather_cat_kernel<<<M, C, 0, stream>>>(h, idx, cat);

  // graph-conv MLP (2C -> C) + stats, BN + GELU
  gemm_bias_stats_kernel<256><<<M / 64, 128, 0, stream>>>(cat, wgh, bg, pre, sum2, sq2, M);
  bn_finalize_kernel<<<1, C, 0, stream>>>(sum2, sq2, mean2, rstd2, 1.f / M);
  bn_gelu_kernel<<<(M * C) / 256, 256, 0, stream>>>(pre, mean2, rstd2, gg, beg, hgh);

  // fc2 + stats, BN + residual
  gemm_bias_stats_kernel<128><<<M / 64, 128, 0, stream>>>(hgh, w2h, b2, pre, sum3, sq3, M);
  bn_finalize_kernel<<<1, C, 0, stream>>>(sum3, sq3, mean3, rstd3, 1.f / M);
  bn_res_kernel<<<(M * C) / 256, 256, 0, stream>>>(pre, mean3, rstd3, g2, be2, x, (float*)d_out);
}
